// Capsule_35811437314062
// MI455X (gfx1250) — matrix operand order repaired and checked by reading, not compiled, not run on hardware
//
#include <hip/hip_runtime.h>
#include <hip/hip_bf16.h>
#include <math.h>

// ---------------------------------------------------------------------------
// Capsule routing for MI455X (gfx1250, wave32, WMMA).
//   hat = x @ kernel        : (128*2048, 128) x (128, 512) -> 512 MB fp32
//   3x dynamic routing      : softmax over 32 capsules, two einsums fused
// GEMM uses bf16-split (hi+lo) WMMA: 3x v_wmma_f32_16x16x32_bf16 per K-step
// for ~fp32 accuracy at bf16 matrix rates (fp32 WMMA would be 8x slower per
// the 16x16x4 vs 16x16x32 K-depth ratio and exceed the ~90us HBM floor).
// ---------------------------------------------------------------------------

typedef __attribute__((ext_vector_type(16))) __bf16 v16bf;
typedef __attribute__((ext_vector_type(8)))  __bf16 v8bf;
typedef __attribute__((ext_vector_type(8)))  float  v8f;

static constexpr int B_ = 128;
static constexpr int N_ = 2048;
static constexpr int K_ = 128;           // input dim
static constexpr int C_ = 32;            // num capsules
static constexpr int D_ = 16;            // dim capsule
static constexpr int M_ = C_ * D_;       // 512 output cols
static constexpr int ROWS = B_ * N_;     // 262144 GEMM rows

// GEMM tiling
static constexpr int TR  = 64;           // rows per block
static constexpr int TM  = 256;          // cols per block
static constexpr int LDK = 136;          // padded K stride (272B rows: 16B-aligned,
                                         // staggers LDS banks)

// routing tiling
static constexpr int NCHUNK = 8;         // blocks per batch image
static constexpr int NPER   = N_ / NCHUNK; // 256 n-rows per block

// ---------------------------------------------------------------------------
// GEMM: hat[g, m] = sum_k x[g,k] * w[k,m], bf16-split via WMMA.
// ---------------------------------------------------------------------------
__global__ __launch_bounds__(256)
void gemm_hat(const float* __restrict__ x, const float* __restrict__ w,
              float* __restrict__ hat)
{
    extern __shared__ __bf16 sm[];
    __bf16* aHi = sm;                    // [TR][LDK]
    __bf16* aLo = aHi + TR * LDK;
    __bf16* bHi = aLo + TR * LDK;        // [TM][LDK]  (kernel, transposed)
    __bf16* bLo = bHi + TM * LDK;

    const int tid    = threadIdx.x;
    const int rowBlk = blockIdx.x >> 1;
    const int colBlk = blockIdx.x & 1;
    const int g0     = rowBlk * TR;
    const int m0     = colBlk * TM;

    // ---- stage A tile (TR x K) as hi/lo bf16, float4 global loads ----
    const float4* x4 = (const float4*)(x + (size_t)g0 * K_);
    for (int i = tid; i < TR * (K_ / 4); i += 256) {
        const int r  = i >> 5;           // K_/4 == 32
        const int kq = i & 31;
        float4 v = x4[r * (K_ / 4) + kq];
        float f[4] = {v.x, v.y, v.z, v.w};
        #pragma unroll
        for (int j = 0; j < 4; ++j) {
            __bf16 h = (__bf16)f[j];
            aHi[r * LDK + kq * 4 + j] = h;
            aLo[r * LDK + kq * 4 + j] = (__bf16)(f[j] - (float)h);
        }
    }
    // ---- stage B tile (K x TM), transposed to [mm][k] ----
    const float4* w4 = (const float4*)w;
    for (int i = tid; i < K_ * (TM / 4); i += 256) {
        const int k  = i >> 6;           // TM/4 == 64
        const int mq = i & 63;
        float4 v = w4[k * (M_ / 4) + (m0 / 4) + mq];
        float f[4] = {v.x, v.y, v.z, v.w};
        #pragma unroll
        for (int j = 0; j < 4; ++j) {
            const int mm = mq * 4 + j;
            __bf16 h = (__bf16)f[j];
            bHi[mm * LDK + k] = h;
            bLo[mm * LDK + k] = (__bf16)(f[j] - (float)h);
        }
    }
    __syncthreads();

    const int wave  = tid >> 5;
    const int lane  = tid & 31;
    const int lhalf = (lane < 16) ? 0 : 1;
    const int l16   = lane & 15;

    // 64 16x16 tiles per block, 8 per wave
    for (int t = 0; t < 8; ++t) {
        const int tile = wave + (t << 3);
        const int r0 = (tile >> 4) * 16; // n-subtile
        const int c0 = (tile & 15) * 16; // m-subtile
        v8f acc = {};
        #pragma unroll
        for (int ks = 0; ks < 4; ++ks) {
            const int kb = ks * 32;
            // A 16x32 bf16 layout: lanes 0-15 row M=l16, K 0..7 & 16..23;
            //                      lanes 16-31           K 8..15 & 24..31.
            const int kA = kb + lhalf * 8;
            // B 32x16 bf16 layout: lanes 0-15 col N=l16, K 0..7 & 16..23;
            //                      lanes 16-31           K 8..15 & 24..31.
            const int kB = kb + lhalf * 8;
            const __bf16* ah = &aHi[(r0 + l16) * LDK + kA];
            const __bf16* al = &aLo[(r0 + l16) * LDK + kA];
            const __bf16* bh = &bHi[(c0 + l16) * LDK + kB];
            const __bf16* bl = &bLo[(c0 + l16) * LDK + kB];
            v8bf a0  = *(const v8bf*)(ah);
            v8bf a1  = *(const v8bf*)(ah + 16);
            v8bf al0 = *(const v8bf*)(al);
            v8bf al1 = *(const v8bf*)(al + 16);
            v8bf b0  = *(const v8bf*)(bh);
            v8bf b1  = *(const v8bf*)(bh + 16);
            v8bf bl0 = *(const v8bf*)(bl);
            v8bf bl1 = *(const v8bf*)(bl + 16);
            v16bf A  = __builtin_shufflevector(a0,  a1,  0,1,2,3,4,5,6,7,8,9,10,11,12,13,14,15);
            v16bf Al = __builtin_shufflevector(al0, al1, 0,1,2,3,4,5,6,7,8,9,10,11,12,13,14,15);
            v16bf Bh = __builtin_shufflevector(b0,  b1,  0,1,2,3,4,5,6,7,8,9,10,11,12,13,14,15);
            v16bf Bl = __builtin_shufflevector(bl0, bl1, 0,1,2,3,4,5,6,7,8,9,10,11,12,13,14,15);
            // a*b ~= ah*bh + ah*bl + al*bh   (drop al*bl)
            acc = __builtin_amdgcn_wmma_f32_16x16x32_bf16(false, A,  false, Bh, (short)0, acc, false, false);
            acc = __builtin_amdgcn_wmma_f32_16x16x32_bf16(false, A,  false, Bl, (short)0, acc, false, false);
            acc = __builtin_amdgcn_wmma_f32_16x16x32_bf16(false, Al, false, Bh, (short)0, acc, false, false);
        }
        // D layout: VGPR i -> row r0 + lhalf*8 + i, col c0 + l16
        float* out = hat + (size_t)(g0 + r0 + lhalf * 8) * M_ + (m0 + c0 + l16);
        #pragma unroll
        for (int i = 0; i < 8; ++i) out[(size_t)i * M_] = acc[i];
    }
}

// ---------------------------------------------------------------------------
// Fused routing pass: for each (b, n): bv[c] = <o[b,c,:], hat[b,n,c,:]>;
// c = softmax_c(bv); accumulate c[c]*hat[b,n,c,:] into per-block partial o.
// o == 0 gives the iteration-0 uniform softmax (1/32) for free.
// Deterministic two-stage reduction (registers -> LDS -> per-block global).
// ---------------------------------------------------------------------------
__global__ __launch_bounds__(256)
void routing(const float* __restrict__ hat, const float* __restrict__ o,
             float* __restrict__ pblk)
{
    __shared__ float o_s[M_];
    __shared__ float wacc[8 * M_];

    const int tid   = threadIdx.x;
    const int b     = blockIdx.x / NCHUNK;
    const int chunk = blockIdx.x % NCHUNK;

    o_s[tid]       = o[b * M_ + tid];
    o_s[tid + 256] = o[b * M_ + tid + 256];
    __syncthreads();

    const int wave = tid >> 5;
    const int lane = tid & 31;        // lane == capsule index c

    float oc[16];
    #pragma unroll
    for (int d = 0; d < 16; ++d) oc[d] = o_s[lane * 16 + d];
    float racc[16];
    #pragma unroll
    for (int d = 0; d < 16; ++d) racc[d] = 0.f;

    const size_t rowBase = ((size_t)b * N_ + (size_t)chunk * NPER) * M_;
    for (int i = 0; i < NPER / 8; ++i) {           // 32 n-rows per wave
        const int n_local = wave * (NPER / 8) + i;
        const float4* hp =
            (const float4*)(hat + rowBase + (size_t)n_local * M_ + lane * 16);
        float4 h0 = hp[0], h1 = hp[1], h2 = hp[2], h3 = hp[3];
        float hv[16] = {h0.x,h0.y,h0.z,h0.w, h1.x,h1.y,h1.z,h1.w,
                        h2.x,h2.y,h2.z,h2.w, h3.x,h3.y,h3.z,h3.w};
        float bc = 0.f;
        #pragma unroll
        for (int d = 0; d < 16; ++d) bc += oc[d] * hv[d];
        // wave32 softmax over capsules
        float m = bc;
        #pragma unroll
        for (int off = 16; off >= 1; off >>= 1) m = fmaxf(m, __shfl_xor(m, off));
        float e = __expf(bc - m);
        float s = e;
        #pragma unroll
        for (int off = 16; off >= 1; off >>= 1) s += __shfl_xor(s, off);
        const float cw = e / s;
        #pragma unroll
        for (int d = 0; d < 16; ++d) racc[d] += cw * hv[d];
    }

    #pragma unroll
    for (int d = 0; d < 16; ++d) wacc[wave * M_ + lane * 16 + d] = racc[d];
    __syncthreads();

    for (int j = tid; j < M_; j += 256) {
        float s = 0.f;
        #pragma unroll
        for (int w8 = 0; w8 < 8; ++w8) s += wacc[w8 * M_ + j];
        pblk[(size_t)blockIdx.x * M_ + j] = s;
    }
}

// ---------------------------------------------------------------------------
// Reduce NCHUNK per-block partials and apply squash. One thread per (b,c).
// ---------------------------------------------------------------------------
__global__ __launch_bounds__(256)
void squash_reduce(const float* __restrict__ pblk, float* __restrict__ out)
{
    const int idx = blockIdx.x * 256 + threadIdx.x;   // b*32 + c
    if (idx >= B_ * C_) return;
    const int b = idx >> 5;
    const int c = idx & 31;

    float v[16];
    #pragma unroll
    for (int d = 0; d < 16; ++d) v[d] = 0.f;
    for (int ch = 0; ch < NCHUNK; ++ch) {
        const float* p = pblk + ((size_t)(b * NCHUNK + ch)) * M_ + c * 16;
        #pragma unroll
        for (int d = 0; d < 16; ++d) v[d] += p[d];
    }
    float s2 = 1e-7f;
    #pragma unroll
    for (int d = 0; d < 16; ++d) s2 += v[d] * v[d];
    const float scale = sqrtf(s2) / (0.5f + s2);
    float* op = out + (size_t)idx * 16;
    #pragma unroll
    for (int d = 0; d < 16; ++d) op[d] = scale * v[d];
}

__global__ __launch_bounds__(256)
void zero_f(float* __restrict__ p, int n)
{
    const int i = blockIdx.x * 256 + threadIdx.x;
    if (i < n) p[i] = 0.f;
}

// ---------------------------------------------------------------------------
extern "C" void kernel_launch(void* const* d_in, const int* in_sizes, int n_in,
                              void* d_out, int out_size, void* d_ws, size_t ws_size,
                              hipStream_t stream)
{
    (void)in_sizes; (void)n_in; (void)out_size; (void)ws_size;
    const float* x = (const float*)d_in[0];   // (128, 2048, 128) fp32
    const float* w = (const float*)d_in[1];   // (128, 512) fp32
    float* out = (float*)d_out;               // (128, 32, 16) fp32

    char* ws = (char*)d_ws;
    float* hat = (float*)ws;                                   // 512 MB
    const size_t hatBytes  = (size_t)ROWS * M_ * sizeof(float);
    float* pblk = (float*)(ws + hatBytes);                     // 2 MB
    const size_t pblkBytes = (size_t)B_ * NCHUNK * M_ * sizeof(float);
    float* o = (float*)(ws + hatBytes + pblkBytes);            // 256 KB

    // o = 0 -> first routing pass sees softmax(0) = uniform 1/32
    zero_f<<<(B_ * M_ + 255) / 256, 256, 0, stream>>>(o, B_ * M_);

    const size_t lds = (size_t)(2 * TR * LDK + 2 * TM * LDK) * sizeof(__bf16); // 174080 B
    gemm_hat<<<(ROWS / TR) * 2, 256, lds, stream>>>(x, w, hat);

    for (int it = 0; it < 3; ++it) {
        routing<<<B_ * NCHUNK, 256, 0, stream>>>(hat, o, pblk);
        squash_reduce<<<(B_ * C_ + 255) / 256, 256, 0, stream>>>(
            pblk, (it == 2) ? out : o);
    }
}
